// RNN_30107720745579
// MI455X (gfx1250) — matrix operand order repaired and checked by reading, not compiled, not run on hardware
//
#include <hip/hip_runtime.h>
#include <math.h>

typedef __attribute__((ext_vector_type(16))) _Float16 v16h;
typedef __attribute__((ext_vector_type(8)))  _Float16 v8h;
typedef __attribute__((ext_vector_type(8)))  float    v8f;

#define T_DIM 1024
#define B_DIM 64
#define C_DIM 512
#define H_DIM 512
#define L_DIM 2
#define M_DIM (T_DIM * B_DIM)   // 65536
#define SCAN_BLOCKS 16

// ---------------------------------------------------------------------------
// WMMA fragment loaders from pre-converted f16 (layouts per CDNA5 ISA §7.12.2)
// ---------------------------------------------------------------------------

// A fragment 16x32 (MxK), row-major f16 source.
// lane<16 : elements = K kb..kb+7  then kb+16..kb+23   (two 16B loads)
// lane>=16: elements = K kb+8..+15 then kb+24..kb+31
__device__ __forceinline__ v16h a_frag_f16(const _Float16* __restrict__ A, int m0,
                                           int kb, int ld, int lane) {
  const int m    = m0 + (lane & 15);
  const int half = lane >> 4;
  const _Float16* p = A + (size_t)m * ld + kb + half * 8;
  v8h lo = *(const v8h*)p;
  v8h hi = *(const v8h*)(p + 16);
  return __builtin_shufflevector(lo, hi, 0, 1, 2, 3, 4, 5, 6, 7,
                                 8, 9, 10, 11, 12, 13, 14, 15);
}

// B fragment 32x16 (KxN) fed as W rows (W is (out,in)=(N,K) row-major =>
// D[m,n] = sum_k A[m,k] * W[n,k]).
// lane<16: W[n0+n][kb..kb+7, kb+16..kb+23]; lane>=16: W[n0+n][kb+8..kb+15, kb+24..kb+31]  (two 16B loads)
__device__ __forceinline__ v16h b_frag_f16(const _Float16* __restrict__ W, int n0,
                                           int kb, int ld, int lane) {
  const int n = n0 + (lane & 15);
  const _Float16* p = W + (size_t)n * ld + kb + (lane >> 4) * 8;
  v8h lo = *(const v8h*)p;
  v8h hi = *(const v8h*)(p + 16);
  return __builtin_shufflevector(lo, hi, 0, 1, 2, 3, 4, 5, 6, 7,
                                 8, 9, 10, 11, 12, 13, 14, 15);
}

__device__ __forceinline__ v8f wmma16(v16h a, v16h b, v8f c) {
  return __builtin_amdgcn_wmma_f32_16x16x32_f16(false, a, false, b,
                                                (short)0, c, false, false);
}

// ---------------------------------------------------------------------------
// Grid-wide barrier (atomic generation counter; SCAN_BLOCKS co-resident blocks)
// ---------------------------------------------------------------------------
__device__ __forceinline__ void grid_sync(unsigned* cnt, volatile unsigned* gen,
                                          unsigned nblocks) {
  __threadfence();   // publish this thread's stores at device scope
  __syncthreads();
  if (threadIdx.x == 0) {
    unsigned g = *gen;
    if (atomicAdd(cnt, 1u) == nblocks - 1u) {
      atomicExch(cnt, 0u);       // reset before release
      __threadfence();
      atomicAdd((unsigned*)gen, 1u);
    } else {
      while (*gen == g) __builtin_amdgcn_s_sleep(1);
    }
    __threadfence();
  }
  __syncthreads();
}

// ---------------------------------------------------------------------------
// Kernel 0: f32 -> f16 conversion (memory-bound, vectorized 4-wide)
// ---------------------------------------------------------------------------
__global__ __launch_bounds__(256)
void cvt_f32_to_f16(const float* __restrict__ src, _Float16* __restrict__ dst,
                    int n) {
  const int i = 4 * (blockIdx.x * 256 + threadIdx.x);
  if (i < n) {
    const float4 v = *(const float4*)(src + i);
    v8h* o = (v8h*)0;  // silence unused-type warnings path; direct stores below
    (void)o;
    _Float16 h0 = (_Float16)v.x, h1 = (_Float16)v.y;
    _Float16 h2 = (_Float16)v.z, h3 = (_Float16)v.w;
    // one 8-byte store
    struct alignas(8) H4 { _Float16 a, b, c, d; };
    *(H4*)(dst + i) = H4{h0, h1, h2, h3};
  }
}

// ---------------------------------------------------------------------------
// Kernel 1: XP = A16[M,C] . WI16[H,C]^T + bi   (all-f16 operands, f32 accum)
// Wave owns a 32x64 tile: 2 M-tiles x 4 N-tiles; the 8 waves of a block share
// m0 and cover all of N, so A fragments hit the WGP$ after the first wave.
// ---------------------------------------------------------------------------
__global__ __launch_bounds__(256)
void xproj_gemm(const _Float16* __restrict__ A16, const _Float16* __restrict__ WI16,
                const float* __restrict__ bi, float* __restrict__ XP) {
  const int lane = threadIdx.x & 31;
  const int wave = threadIdx.x >> 5;
  const int m0   = blockIdx.x * 32;
  const int n0   = wave * 64;

  v8f zero = {};
  v8f acc[2][4];
#pragma unroll
  for (int i = 0; i < 2; ++i)
#pragma unroll
    for (int j = 0; j < 4; ++j) acc[i][j] = zero;

  for (int kb = 0; kb < C_DIM; kb += 32) {
    v16h a0 = a_frag_f16(A16, m0,      kb, C_DIM, lane);
    v16h a1 = a_frag_f16(A16, m0 + 16, kb, C_DIM, lane);
#pragma unroll
    for (int j = 0; j < 4; ++j) {
      v16h b = b_frag_f16(WI16, n0 + 16 * j, kb, C_DIM, lane);
      acc[0][j] = wmma16(a0, b, acc[0][j]);
      acc[1][j] = wmma16(a1, b, acc[1][j]);
    }
  }

  const int nlo  = lane & 15;
  const int mrow = (lane >> 4) * 8;
#pragma unroll
  for (int j = 0; j < 4; ++j) {
    const int col  = n0 + 16 * j + nlo;
    const float bb = bi[col];
#pragma unroll
    for (int i = 0; i < 2; ++i) {
      float* o = XP + (size_t)(m0 + 16 * i + mrow) * H_DIM + col;
#pragma unroll
      for (int r = 0; r < 8; ++r)
        o[(size_t)r * H_DIM] = acc[i][j][r] + bb;
    }
  }
}

// ---------------------------------------------------------------------------
// Kernel 2: persistent recurrent scan.
// 16 blocks x 8 waves. Block owns 32 hidden columns; wave tile = (wave&3)*16
// batch rows x 16 cols. Wh slice lives in VGPRs as 16 f16 B-fragments (full
// K=512). h state ping-pongs between two f16 buffers (converted once at
// production); per-step path: 16 f16 A-loads + 16 WMMA + tanh + grid barrier.
// ---------------------------------------------------------------------------
__global__ __launch_bounds__(256)
void rnn_scan(const float* __restrict__ XP, const _Float16* __restrict__ WH16,
              const float* __restrict__ bh, const float* __restrict__ h0,
              float* __restrict__ out, float* __restrict__ hn,
              _Float16* __restrict__ hf0, _Float16* __restrict__ hf1,
              unsigned* __restrict__ sync) {
  const int lane = threadIdx.x & 31;
  const int wave = threadIdx.x >> 5;
  const int m0   = (wave & 3) * 16;                       // batch-row tile
  const int n0   = blockIdx.x * 32 + (wave >> 2) * 16;    // hidden-col tile

  // init: h0 (f32) -> hf0 (f16), grid-stride disjoint
  for (int i = blockIdx.x * blockDim.x + threadIdx.x; i < B_DIM * H_DIM;
       i += SCAN_BLOCKS * blockDim.x)
    hf0[i] = (_Float16)h0[i];

  // register-resident Wh fragments for this wave's 16 columns, all of K
  v16h bw[16];
#pragma unroll
  for (int k = 0; k < 16; ++k)
    bw[k] = b_frag_f16(WH16, n0, k * 32, H_DIM, lane);

  const int nlo    = lane & 15;
  const int mrow   = (lane >> 4) * 8;
  const int col    = n0 + nlo;
  const float bias = bh[col];
  const v8f zero   = {};

  grid_sync(sync, sync + 1, SCAN_BLOCKS);

  for (int t = 0; t < T_DIM; ++t) {
    const _Float16* hc  = (t & 1) ? hf1 : hf0;
    _Float16*       hnx = (t & 1) ? hf0 : hf1;

    v8f acc = zero;
#pragma unroll
    for (int k = 0; k < 16; ++k) {
      v16h a = a_frag_f16(hc, m0, k * 32, H_DIM, lane);
      acc = wmma16(a, bw[k], acc);
    }

    const size_t tb = (size_t)t * (B_DIM * H_DIM);
#pragma unroll
    for (int r = 0; r < 8; ++r) {
      const int row    = m0 + mrow + r;
      const size_t hix = (size_t)row * H_DIM + col;
      const float v    = tanhf(acc[r] + XP[tb + hix] + bias);
      out[tb + hix] = v;
      hnx[hix]      = (_Float16)v;     // convert once at production
      if (t == T_DIM - 1) hn[hix] = v;
    }

    if (t + 1 < T_DIM)  // pull next timestep's xproj slice toward L2/L0
      __builtin_prefetch(XP + tb + (size_t)(B_DIM * H_DIM) +
                         (size_t)(m0 + mrow) * H_DIM + col, 0, 1);

    grid_sync(sync, sync + 1, SCAN_BLOCKS);
  }
}

// ---------------------------------------------------------------------------
// Host entry
// ---------------------------------------------------------------------------
extern "C" void kernel_launch(void* const* d_in, const int* in_sizes, int n_in,
                              void* d_out, int out_size, void* d_ws, size_t ws_size,
                              hipStream_t stream) {
  (void)in_sizes; (void)n_in; (void)out_size; (void)ws_size;
  const float* X    = (const float*)d_in[0];   // [T,B,C]
  const float* h0   = (const float*)d_in[1];   // [L,B,H]
  const float* W_ih = (const float*)d_in[2];   // [L,H,C]
  const float* b_ih = (const float*)d_in[3];   // [L,H]
  const float* W_hh = (const float*)d_in[4];   // [L,H,H]
  const float* b_hh = (const float*)d_in[5];   // [L,H]

  float* out = (float*)d_out;                               // [T,B,H]
  float* hn  = out + (size_t)T_DIM * B_DIM * H_DIM;         // [L,B,H]

  // workspace layout
  float*     XP   = (float*)d_ws;                           // [M,H]  f32  128 MB
  _Float16*  A16  = (_Float16*)(XP + (size_t)M_DIM * H_DIM);// [M,C]  f16   64 MB
  _Float16*  WI16 = A16 + (size_t)M_DIM * C_DIM;            // [H,C]  f16  512 KB
  _Float16*  WH16 = WI16 + (size_t)H_DIM * C_DIM;           // [H,H]  f16  512 KB
  _Float16*  hf0  = WH16 + (size_t)H_DIM * H_DIM;           // [B,H]  f16   64 KB
  _Float16*  hf1  = hf0 + B_DIM * H_DIM;                    // [B,H]  f16   64 KB
  unsigned*  sync = (unsigned*)(hf1 + B_DIM * H_DIM);

  hipMemsetAsync(sync, 0, 2 * sizeof(unsigned), stream);

  const int nA = M_DIM * C_DIM;      // 33.5M
  const int nW = H_DIM * C_DIM;      // 256K

  for (int l = 0; l < L_DIM; ++l) {
    const float* In = (l == 0) ? X : out;   // layer-1 input = layer-0 output
    cvt_f32_to_f16<<<nA / 4 / 256, 256, 0, stream>>>(In, A16, nA);
    cvt_f32_to_f16<<<nW / 4 / 256, 256, 0, stream>>>(
        W_ih + (size_t)l * H_DIM * C_DIM, WI16, nW);
    cvt_f32_to_f16<<<nW / 4 / 256, 256, 0, stream>>>(
        W_hh + (size_t)l * H_DIM * H_DIM, WH16, nW);

    xproj_gemm<<<M_DIM / 32, 256, 0, stream>>>(A16, WI16,
                                               b_ih + (size_t)l * H_DIM, XP);
    rnn_scan<<<SCAN_BLOCKS, 256, 0, stream>>>(
        XP, WH16, b_hh + (size_t)l * H_DIM, h0 + (size_t)l * B_DIM * H_DIM,
        out, hn + (size_t)l * B_DIM * H_DIM, hf0, hf1, sync);
  }
}
